// GraphCPI_GCN_36850819400362
// MI455X (gfx1250) — matrix operand order repaired and checked by reading, not compiled, not run on hardware
//
#include <hip/hip_runtime.h>
#include <hip/hip_bf16.h>

typedef __attribute__((ext_vector_type(16))) _Float16 v16h;
typedef __attribute__((ext_vector_type(8)))  float    v8f;

#define NNODES 65536
#define NPG    64
#define BSZ    1024
#define SEQL   1000
#define EMBDIM 100
#define CONVO  32
#define CONVH  93      // 100 - 8 + 1
#define CONVK  8000    // 1000 channels * 8 taps
#define CONVKT 250     // 8000 / 32

// ---------------------------------------------------------------------------
// Fragment packing: pre-swizzle operands into the per-lane WMMA element order
// (ISA 7.12.2) so the GEMM inner loop does aligned 32B per-lane loads only.
//   A frag element (mt,kt,lane,e): row = mt*16+(lane&15),
//                                  k   = kt*32+(lane>>4)*8 + e + (e>=8 ? 8:0)
//   B frag element (nt,kt,lane,e): col = nt*16+(lane&15),
//                                  k   = kt*32+(lane>>4)*8 + e + (e>=8 ? 8:0)
// Zero-padded outside [M,K]/[K,N], so the GEMM needs no bounds checks on K.
// ---------------------------------------------------------------------------
__global__ void k_packA(const float* __restrict__ A, _Float16* __restrict__ Ap,
                        int M, int K, int Kt, long total) {
    long i = (long)blockIdx.x * blockDim.x + threadIdx.x;
    if (i >= total) return;
    int e    = (int)(i & 15);
    int lane = (int)((i >> 4) & 31);
    long tk  = i >> 9;                       // mt*Kt + kt
    int kt   = (int)(tk % Kt);
    int mt   = (int)(tk / Kt);
    int row  = mt * 16 + (lane & 15);
    int k    = kt * 32 + (lane >> 4) * 8 + e + ((e >= 8) ? 8 : 0);
    float v  = (row < M && k < K) ? A[(long)row * K + k] : 0.0f;
    Ap[i] = (_Float16)v;
}
__global__ void k_packB(const float* __restrict__ B, _Float16* __restrict__ Bp,
                        int N, int K, int Kt, long total) {
    long i = (long)blockIdx.x * blockDim.x + threadIdx.x;
    if (i >= total) return;
    int e    = (int)(i & 15);
    int lane = (int)((i >> 4) & 31);
    long tk  = i >> 9;                       // nt*Kt + kt
    int kt   = (int)(tk % Kt);
    int nt   = (int)(tk / Kt);
    int col  = nt * 16 + (lane & 15);
    int k    = kt * 32 + (lane >> 4) * 8 + e + ((e >= 8) ? 8 : 0);
    float v  = (k < K && col < N) ? B[(long)k * N + col] : 0.0f;
    Bp[i] = (_Float16)v;
}
// B given transposed as [N,K] row-major (conv weights [32, 8000])
__global__ void k_packBT(const float* __restrict__ Bt, _Float16* __restrict__ Bp,
                         int N, int K, int Kt, long total) {
    long i = (long)blockIdx.x * blockDim.x + threadIdx.x;
    if (i >= total) return;
    int e    = (int)(i & 15);
    int lane = (int)((i >> 4) & 31);
    long tk  = i >> 9;
    int kt   = (int)(tk % Kt);
    int nt   = (int)(tk / Kt);
    int col  = nt * 16 + (lane & 15);
    int k    = kt * 32 + (lane >> 4) * 8 + e + ((e >= 8) ? 8 : 0);
    float v  = (k < K && col < N) ? Bt[(long)col * K + k] : 0.0f;
    Bp[i] = (_Float16)v;
}

// ---------------------------------------------------------------------------
// Branch-free packed GEMM: one wave per 16x16 C tile; inner loop = 2 aligned
// 32B fragment loads + 1 WMMA, unrolled x4 so loads batch under one wait.
// C/D layout: VGPR j -> row = 16*mt + (lane>>4)*8 + j, col = 16*nt + (lane&15)
// ---------------------------------------------------------------------------
__global__ __launch_bounds__(256)
void wmma_gemm_packed(const _Float16* __restrict__ Ap, const _Float16* __restrict__ Bp,
                      const float* __restrict__ bias, float* __restrict__ C,
                      int M, int N, int Kt, int relu)
{
    const int lane   = threadIdx.x & 31;
    const int wid    = threadIdx.x >> 5;
    const int ntiles = (N + 15) >> 4;
    const int mtiles = (M + 15) >> 4;
    long tile = (long)blockIdx.x * (blockDim.x >> 5) + wid;
    if (tile >= (long)mtiles * ntiles) return;        // wave-uniform exit
    const int mt = (int)(tile / ntiles);
    const int nt = (int)(tile % ntiles);

    const v16h* __restrict__ Af = ((const v16h*)Ap) + ((long)mt * Kt * 32 + lane);
    const v16h* __restrict__ Bf = ((const v16h*)Bp) + ((long)nt * Kt * 32 + lane);

    v8f acc = {};
#pragma unroll 4
    for (int kt = 0; kt < Kt; ++kt) {
        __builtin_prefetch((const void*)(Af + (long)(kt + 8) * 32), 0, 0);
        v16h a = Af[(long)kt * 32];
        v16h b = Bf[(long)kt * 32];
        acc = __builtin_amdgcn_wmma_f32_16x16x32_f16(false, a, false, b,
                                                     (short)0, acc, false, false);
    }

    const int c_col = nt * 16 + (lane & 15);
    if (c_col < N) {
        float bv = bias ? bias[c_col] : 0.0f;
        const int r0 = mt * 16 + (lane >> 4) * 8;
#pragma unroll
        for (int j = 0; j < 8; ++j) {
            int r = r0 + j;
            if (r < M) {
                float v = acc[j] + bv;
                if (relu) v = fmaxf(v, 0.0f);
                C[(long)r * N + c_col] = v;
            }
        }
    }
}

// ---------------------------------------------------------------------------
// Protein conv1d as implicit gather-GEMM (M=BSZ*93, N=32, K=8000, exact tiles):
//   out[(b,h), o] = sum_{i<1000,k<8} emb[target[b,i], h+k] * convw[o, i, k]
// A fragment halves each map to ONE token row, 8 contiguous floats of emb.
// Conv weights pre-packed to f16 fragments (cwp). No masking -> EXEC all-1s.
// Output stored as ct[b, o*93 + h] to match reshape(B, 32*93).
// ---------------------------------------------------------------------------
__global__ __launch_bounds__(256)
void wmma_conv_gemm(const float* __restrict__ emb, const int* __restrict__ target,
                    const _Float16* __restrict__ cwp, const float* __restrict__ convb,
                    float* __restrict__ ct)
{
    const int lane = threadIdx.x & 31;
    const int wid  = threadIdx.x >> 5;
    long tile = (long)blockIdx.x * (blockDim.x >> 5) + wid;
    const long total = (long)((BSZ * CONVH) / 16) * 2;    // 5952 m-tiles * 2 n-tiles
    if (tile >= total) return;
    const int mt = (int)(tile >> 1);
    const int nt = (int)(tile & 1);

    const int r  = mt * 16 + (lane & 15);     // row = b*93 + h
    const int bb = r / CONVH;
    const int hh = r % CONVH;
    const int a_koff = (lane >> 4) * 8;       // multiple of 8 -> whole tap group
    const int oc     = nt * 16 + (lane & 15);
    const int*  trow = target + bb * SEQL;
    const v16h* __restrict__ Bf = ((const v16h*)cwp) + ((long)nt * CONVKT * 32 + lane);

    v8f acc = {};
#pragma unroll 2
    for (int k0 = 0; k0 < CONVK; k0 += 32) {
        const int i0 = (k0 + a_koff) >> 3;        // seq position for elems 0..7
        const int i1 = i0 + 2;                    // seq position for elems 8..15
        const int tok0 = trow[i0];
        const int tok1 = trow[i1];
        const float* p0 = emb + (long)tok0 * EMBDIM + hh;
        const float* p1 = emb + (long)tok1 * EMBDIM + hh;
        v16h a;
#pragma unroll
        for (int e = 0; e < 8; ++e) {
            a[e]     = (_Float16)p0[e];
            a[e + 8] = (_Float16)p1[e];
        }
        v16h b = Bf[(long)(k0 >> 5) * 32];
        acc = __builtin_amdgcn_wmma_f32_16x16x32_f16(false, a, false, b,
                                                     (short)0, acc, false, false);
    }

    const float bv = convb[oc];
    const int r0 = mt * 16 + (lane >> 4) * 8;
#pragma unroll
    for (int j = 0; j < 8; ++j) {
        int rr = r0 + j;
        int ob = rr / CONVH, oh = rr % CONVH;
        ct[(long)ob * (CONVO * CONVH) + oc * CONVH + oh] = acc[j] + bv;
    }
}

// ------------------------- elementwise / graph kernels ----------------------
__global__ void k_fill1(float* p, long n) {
    long i = (long)blockIdx.x * blockDim.x + threadIdx.x;
    if (i < n) p[i] = 1.0f;                        // self-loop contributes deg 1
}
__global__ void k_deg_accum(const int* __restrict__ dst, float* deg, int E) {
    int i = blockIdx.x * blockDim.x + threadIdx.x;
    if (i < E) (void)unsafeAtomicAdd(&deg[dst[i]], 1.0f);
}
__global__ void k_rsqrt(float* p, long n) {
    long i = (long)blockIdx.x * blockDim.x + threadIdx.x;
    if (i < n) p[i] = rsqrtf(p[i]);
}
// OUT[n,c] = bias[c] + dinv[n]^2 * HW[n,c]   (bias + self-loop message)
__global__ void k_scatter_init(const float* __restrict__ hw, const float* __restrict__ dinv,
                               const float* __restrict__ bias, float* __restrict__ out,
                               int n, int C) {
    long i = (long)blockIdx.x * blockDim.x + threadIdx.x;
    if (i >= (long)n * C) return;
    int node = (int)(i / C), c = (int)(i % C);
    float di = dinv[node];
    out[i] = bias[c] + di * di * hw[i];
}
// one wave per edge; lanes stride channels; native f32 global atomics
__global__ __launch_bounds__(256)
void k_edge_scatter(const int* __restrict__ src, const int* __restrict__ dst,
                    const float* __restrict__ dinv, const float* __restrict__ hw,
                    float* __restrict__ out, int E, int C) {
    int e = blockIdx.x * (blockDim.x >> 5) + (threadIdx.x >> 5);
    if (e >= E) return;
    int lane = threadIdx.x & 31;
    int s = src[e], d = dst[e];
    float nrm = dinv[s] * dinv[d];
    const float* hs = hw + (long)s * C;
    float* od = out + (long)d * C;
    for (int c = lane; c < C; c += 32)
        (void)unsafeAtomicAdd(&od[c], nrm * hs[c]);
}
__global__ void k_relu(float* p, long n) {
    long i = (long)blockIdx.x * blockDim.x + threadIdx.x;
    if (i < n) p[i] = fmaxf(p[i], 0.0f);
}
__global__ void k_pool_max(const float* __restrict__ h, float* __restrict__ pooled, int C) {
    long i = (long)blockIdx.x * blockDim.x + threadIdx.x;
    if (i >= (long)BSZ * C) return;
    int g = (int)(i / C), c = (int)(i % C);
    const float* base = h + ((long)g * NPG) * C + c;
    float m = -3.0e38f;
    for (int k = 0; k < NPG; ++k) m = fmaxf(m, base[(long)k * C]);
    pooled[i] = m;
}
__global__ void k_concat(const float* __restrict__ d, const float* __restrict__ xt,
                         float* __restrict__ hcat) {
    long i = (long)blockIdx.x * blockDim.x + threadIdx.x;
    if (i >= (long)BSZ * 256) return;
    int b = (int)(i >> 8), c = (int)(i & 255);
    hcat[i] = (c < 128) ? d[b * 128 + c] : xt[b * 128 + (c - 128)];
}

// ---------------------------------------------------------------------------
static inline unsigned blk256(long n) { return (unsigned)((n + 255) / 256); }

static void gemm(const float* A, const float* B, const float* bias, float* C,
                 int M, int N, int K, int relu,
                 _Float16* Ap, _Float16* Bp, hipStream_t s) {
    int Kt = (K + 31) / 32;
    int mtiles = (M + 15) / 16, ntiles = (N + 15) / 16;
    long na = (long)mtiles * Kt * 512;
    long nb = (long)ntiles * Kt * 512;
    k_packA<<<blk256(na), 256, 0, s>>>(A, Ap, M, K, Kt, na);
    k_packB<<<blk256(nb), 256, 0, s>>>(B, Bp, N, K, Kt, nb);
    long tiles = (long)mtiles * ntiles;
    wmma_gemm_packed<<<(int)((tiles + 7) / 8), 256, 0, s>>>(Ap, Bp, bias, C,
                                                            M, N, Kt, relu);
}

extern "C" void kernel_launch(void* const* d_in, const int* in_sizes, int n_in,
                              void* d_out, int out_size, void* d_ws, size_t ws_size,
                              hipStream_t stream) {
    (void)n_in; (void)out_size; (void)ws_size;
    const float* x      = (const float*)d_in[0];
    const int*   eidx   = (const int*)d_in[1];
    const int    E      = in_sizes[1] / 2;
    const int*   src    = eidx;
    const int*   dstp   = eidx + E;
    const int*   target = (const int*)d_in[3];
    const float* W1 = (const float*)d_in[4],  *b1 = (const float*)d_in[5];
    const float* W2 = (const float*)d_in[6],  *b2 = (const float*)d_in[7];
    const float* W3 = (const float*)d_in[8],  *b3 = (const float*)d_in[9];
    const float* Wg1= (const float*)d_in[10], *bg1= (const float*)d_in[11];
    const float* Wg2= (const float*)d_in[12], *bg2= (const float*)d_in[13];
    const float* emb  = (const float*)d_in[14];
    const float* convw= (const float*)d_in[15], *convb = (const float*)d_in[16];
    const float* Wxt= (const float*)d_in[17], *bxt= (const float*)d_in[18];
    const float* Wa = (const float*)d_in[19], *ba = (const float*)d_in[20];
    const float* Wb = (const float*)d_in[21], *bbv= (const float*)d_in[22];
    const float* Wc = (const float*)d_in[23], *bc = (const float*)d_in[24];
    float* out = (float*)d_out;

    float* ws = (float*)d_ws;
    const long BIG = (long)NNODES * 312;              // 20,447,232 floats
    float* bufA = ws;                                  // GEMM HW output
    float* bufB = ws + BIG;                            // activations ping-pong
    float* dinv = ws + 2 * BIG;                        // NNODES
    _Float16* Apack = (_Float16*)(ws + 2 * BIG + 65536);       // 10.5M f16
    _Float16* Bpack = (_Float16*)(ws + 2 * BIG + 65536 + 5242880); // 0.52M f16
    _Float16* cwp   = (_Float16*)(ws + 2 * BIG + 65536 + 5242880 + 262144);
    // small buffers carved from bufA once the graph branch no longer needs it
    float* pooled = bufA + 0;        // 1024*312
    float* dm1    = bufA + 400000;   // 1024*1024
    float* dvec   = bufA + 1500000;  // 1024*128
    float* ct     = bufA + 1700000;  // 1024*32*93
    float* xt     = bufA + 4800000;  // 1024*128
    float* hcat   = bufA + 5000000;  // 1024*256
    float* h1     = bufA + 5300000;  // 1024*1024
    float* h2     = bufA + 6400000;  // 1024*512

    // --- symmetric-norm degree: deg = 1 (self loop) + scatter(1 over dst) ---
    k_fill1    <<<blk256(NNODES), 256, 0, stream>>>(dinv, NNODES);
    k_deg_accum<<<blk256(E),      256, 0, stream>>>(dstp, dinv, E);
    k_rsqrt    <<<blk256(NNODES), 256, 0, stream>>>(dinv, NNODES);

    // --- 3 GCN layers: packed WMMA transform + edge scatter-add + ReLU ---
    struct { const float* W; const float* b; int ci, co; } L[3] = {
        {W1, b1, 78, 78}, {W2, b2, 78, 156}, {W3, b3, 156, 312}};
    const float* H = x;
    for (int l = 0; l < 3; ++l) {
        long nc = (long)NNODES * L[l].co;
        gemm(H, L[l].W, nullptr, bufA, NNODES, L[l].co, L[l].ci, 0,
             Apack, Bpack, stream);
        k_scatter_init<<<blk256(nc), 256, 0, stream>>>(bufA, dinv, L[l].b, bufB,
                                                       NNODES, L[l].co);
        k_edge_scatter<<<(E + 7) / 8, 256, 0, stream>>>(src, dstp, dinv, bufA,
                                                        bufB, E, L[l].co);
        k_relu<<<blk256(nc), 256, 0, stream>>>(bufB, nc);
        H = bufB;
    }

    // --- pool + drug MLP ---
    k_pool_max<<<blk256((long)BSZ * 312), 256, 0, stream>>>(bufB, pooled, 312);
    gemm(pooled, Wg1, bg1, dm1, BSZ, 1024, 312, 1, Apack, Bpack, stream);
    gemm(dm1,    Wg2, bg2, dvec, BSZ, 128, 1024, 0, Apack, Bpack, stream);

    // --- protein branch: pack conv weights, conv-as-GEMM, then Wxt ---
    {
        long nb = (long)2 * CONVKT * 512;             // 2 n-tiles, 250 k-tiles
        k_packBT<<<blk256(nb), 256, 0, stream>>>(convw, cwp, CONVO, CONVK,
                                                 CONVKT, nb);
        long tiles = (long)((BSZ * CONVH) / 16) * 2;
        wmma_conv_gemm<<<(int)((tiles + 7) / 8), 256, 0, stream>>>(emb, target,
                                                                   cwp, convb, ct);
    }
    gemm(ct, Wxt, bxt, xt, BSZ, 128, CONVO * CONVH, 0, Apack, Bpack, stream);

    // --- fusion MLP ---
    k_concat<<<blk256((long)BSZ * 256), 256, 0, stream>>>(dvec, xt, hcat);
    gemm(hcat, Wa, ba,  h1, BSZ, 1024, 256, 1, Apack, Bpack, stream);
    gemm(h1,   Wb, bbv, h2, BSZ, 512, 1024, 1, Apack, Bpack, stream);
    gemm(h2,   Wc, bc,  out, BSZ, 1, 512, 0, Apack, Bpack, stream);
}
